// AttentionAggregator_45552423142057
// MI455X (gfx1250) — matrix operand order repaired and checked by reading, not compiled, not run on hardware
//
#include <hip/hip_runtime.h>
#include <hip/hip_bf16.h>

typedef __attribute__((ext_vector_type(16))) __bf16 v16bf;
typedef __attribute__((ext_vector_type(8)))  float  v8f;

#define DIN  256
#define DOUT 128
#define KT   (DIN / 32)     // 8 k-tiles of 32
#define NT   (DOUT / 16)    // 8 n-tiles of 16
// packed W fragment count per matrix: KT * NT * 32 lanes * 16 bf16
#define WPK_ELEMS (KT * NT * 32 * 16)   // 32768

// ---------------------------------------------------------------------------
// Kernel 0: repack W (DIN x DOUT, row-major f32) into wave32 WMMA B-fragment
// order as bf16. B 32x16 k order used here, contiguous like A (per 32-K tile):
//   lane<16 : col N = lane,    element j -> K = j
//   lane>=16: col N = lane-16, element j -> K = 16 + j
// Linear index: (((kt*NT + nt)*32 + lane)*16 + j)
// ---------------------------------------------------------------------------
__global__ void pack_w_kernel(const float* __restrict__ W0,
                              const float* __restrict__ W1,
                              __bf16* __restrict__ w0pk,
                              __bf16* __restrict__ w1pk) {
    int idx = blockIdx.x * blockDim.x + threadIdx.x;   // 0 .. 2*WPK_ELEMS-1
    const float* W = (idx < WPK_ELEMS) ? W0 : W1;
    __bf16* out    = (idx < WPK_ELEMS) ? w0pk : w1pk;
    int e = idx & (WPK_ELEMS - 1);
    int j    = e & 15;
    int lane = (e >> 4) & 31;
    int nt   = (e >> 9) & (NT - 1);
    int kt   = e >> 12;
    int laneLo = lane & 15;
    int K = kt * 32 + ((lane >= 16) ? 16 : 0) + j;
    int Ncol = nt * 16 + laneLo;
    out[e] = (__bf16)W[K * DOUT + Ncol];
}

// ---------------------------------------------------------------------------
// Kernel 1: dual GEMM. vw_self = vecs @ W0, vw_neigh = vecs @ W1.
// Block = 256 threads (8 waves). Block handles 16 rows x full DOUT.
// Wave w handles n-tile w (16 cols). Per k-tile: build A fragment from f32
// rows (cvt to bf16), load packed B fragments, 2x v_wmma_f32_16x16x32_bf16.
// ---------------------------------------------------------------------------
__global__ __launch_bounds__(256)
void gemm_dual_kernel(const float* __restrict__ vecs,
                      const __bf16* __restrict__ w0pk,
                      const __bf16* __restrict__ w1pk,
                      float* __restrict__ vw_self,
                      float* __restrict__ vw_neigh,
                      int N) {
    const int m0   = blockIdx.x * 16;
    const int wid  = threadIdx.x >> 5;          // 0..7 -> n-tile
    const int lane = threadIdx.x & 31;
    const int laneLo = lane & 15;
    const bool hi  = lane >= 16;

    int arow_i = m0 + laneLo;
    if (arow_i >= N) arow_i = N - 1;            // clamp (EXEC stays all-ones)
    const float* arow = vecs + (size_t)arow_i * DIN;

    v8f accS = {};
    v8f accN = {};

    #pragma unroll
    for (int kt = 0; kt < KT; ++kt) {
        // A fragment: 16-bit A 16x32 layout
        const int kb0 = kt * 32 + (hi ? 16 : 0);    // elements 0..7
        const int kb1 = kt * 32 + (hi ? 24 : 8);    // elements 8..15
        v16bf a;
        #pragma unroll
        for (int j = 0; j < 8; ++j) a[j]     = (__bf16)arow[kb0 + j];
        #pragma unroll
        for (int j = 0; j < 8; ++j) a[8 + j] = (__bf16)arow[kb1 + j];

        const size_t boff = ((size_t)(kt * NT + wid) * 32 + lane) * 16;
        v16bf b0 = *(const v16bf*)(w0pk + boff);   // 32B contiguous
        v16bf b1 = *(const v16bf*)(w1pk + boff);

        accS = __builtin_amdgcn_wmma_f32_16x16x32_bf16(
                   false, a, false, b0, (short)0, accS, false, false);
        accN = __builtin_amdgcn_wmma_f32_16x16x32_bf16(
                   false, a, false, b1, (short)0, accN, false, false);
    }

    // C/D layout: lane<16 -> rows 0..7 (VGPR r), lane>=16 -> rows 8..15
    const int col = wid * 16 + laneLo;
    #pragma unroll
    for (int r = 0; r < 8; ++r) {
        int row = m0 + (hi ? 8 + r : r);
        if (row < N) {
            vw_self [(size_t)row * DOUT + col] = accS[r];
            vw_neigh[(size_t)row * DOUT + col] = accN[r];
        }
    }
}

// ---------------------------------------------------------------------------
// Kernel 2: per-node scores. s_self = vw_neigh . att0 + att_b0,
//                            s_neigh = vw_neigh . att1 + att_b1.
// One wave (32 lanes) per node; 4 features per lane; shuffle reduction.
// ---------------------------------------------------------------------------
__global__ __launch_bounds__(256)
void score_kernel(const float* __restrict__ vw_neigh,
                  const float* __restrict__ att0,
                  const float* __restrict__ att1,
                  const float* __restrict__ att_b0,
                  const float* __restrict__ att_b1,
                  float* __restrict__ s_self,
                  float* __restrict__ s_neigh,
                  int N) {
    const int wid  = threadIdx.x >> 5;
    const int lane = threadIdx.x & 31;
    const int node = blockIdx.x * 8 + wid;
    if (node >= N) return;
    const float* vr = vw_neigh + (size_t)node * DOUT + lane * 4;
    float a0 = 0.f, a1 = 0.f;
    #pragma unroll
    for (int j = 0; j < 4; ++j) {
        float v = vr[j];
        a0 += v * att0[lane * 4 + j];
        a1 += v * att1[lane * 4 + j];
    }
    #pragma unroll
    for (int m = 16; m > 0; m >>= 1) {
        a0 += __shfl_xor(a0, m, 32);
        a1 += __shfl_xor(a1, m, 32);
    }
    if (lane == 0) {
        s_self[node]  = a0 + att_b0[0];
        s_neigh[node] = a1 + att_b1[0];
    }
}

// ---------------------------------------------------------------------------
// Kernel 3: per-row softmax-attention aggregation + epilogue.
// adj_rows is sorted -> binary search the row's edge range, then:
//   pass 1: row max of leaky_relu(s_neigh[col] + s_self[row])
//   pass 2: denom = sum exp(e - m);  scale = deg / denom
//   pass 3: chunk edges through LDS (w, col); each thread owns one feature.
// out = relu(msg + b1) + relu(vw_self + b0).  Zero atomics.
// ---------------------------------------------------------------------------
__global__ __launch_bounds__(128)
void aggregate_kernel(const int*   __restrict__ adj_rows,
                      const int*   __restrict__ adj_cols,
                      const float* __restrict__ adj_vals,
                      const float* __restrict__ s_self,
                      const float* __restrict__ s_neigh,
                      const float* __restrict__ vw_neigh,
                      const float* __restrict__ vw_self,
                      const float* __restrict__ b0,
                      const float* __restrict__ b1,
                      float* __restrict__ out,
                      int E) {
    __shared__ int   srange[2];
    __shared__ float sred[128];
    __shared__ float sw[128];
    __shared__ int   sc[128];

    const int row = blockIdx.x;
    const int t   = threadIdx.x;          // feature index 0..127

    if (t == 0) {
        int lo = 0, hi = E;
        while (lo < hi) { int mid = (lo + hi) >> 1; if (adj_rows[mid] < row) lo = mid + 1; else hi = mid; }
        srange[0] = lo;
        hi = E;
        while (lo < hi) { int mid = (lo + hi) >> 1; if (adj_rows[mid] < row + 1) lo = mid + 1; else hi = mid; }
        srange[1] = lo;
    }
    __syncthreads();
    const int start = srange[0];
    const int end   = srange[1];
    const int deg   = end - start;

    float acc = 0.f;
    if (deg > 0) {
        const float ss = s_self[row];
        // pass 1: max
        float lm = -3.402823466e38f;
        for (int j = start + t; j < end; j += 128) {
            float e = s_neigh[adj_cols[j]] + ss;
            e = (e > 0.f) ? e : 0.2f * e;
            lm = fmaxf(lm, e);
        }
        sred[t] = lm; __syncthreads();
        #pragma unroll
        for (int s = 64; s > 0; s >>= 1) {
            if (t < s) sred[t] = fmaxf(sred[t], sred[t + s]);
            __syncthreads();
        }
        const float m = sred[0];
        __syncthreads();
        // pass 2: denom
        float ls = 0.f;
        for (int j = start + t; j < end; j += 128) {
            float e = s_neigh[adj_cols[j]] + ss;
            e = (e > 0.f) ? e : 0.2f * e;
            ls += __expf(e - m);
        }
        sred[t] = ls; __syncthreads();
        #pragma unroll
        for (int s = 64; s > 0; s >>= 1) {
            if (t < s) sred[t] = sred[t] + sred[t + s];
            __syncthreads();
        }
        const float scale = (float)deg / sred[0];
        __syncthreads();
        // pass 3: chunked weighted accumulation
        for (int c = start; c < end; c += 128) {
            const int n = min(128, end - c);
            if (t < n) {
                const int j   = c + t;
                const int col = adj_cols[j];
                float e = s_neigh[col] + ss;
                e = (e > 0.f) ? e : 0.2f * e;
                sw[t] = adj_vals[j] * __expf(e - m) * scale;
                sc[t] = col;
            }
            __syncthreads();
            for (int j = 0; j < n; ++j)
                acc += sw[j] * vw_neigh[(size_t)sc[j] * DOUT + t];
            __syncthreads();
        }
    }

    const float o = fmaxf(acc + b1[t], 0.f)
                  + fmaxf(vw_self[(size_t)row * DOUT + t] + b0[t], 0.f);
    out[(size_t)row * DOUT + t] = o;
}

// ---------------------------------------------------------------------------
extern "C" void kernel_launch(void* const* d_in, const int* in_sizes, int n_in,
                              void* d_out, int out_size, void* d_ws, size_t ws_size,
                              hipStream_t stream) {
    const float* vecs     = (const float*)d_in[0];
    const float* adj_vals = (const float*)d_in[1];
    const float* W0       = (const float*)d_in[2];
    const float* W1       = (const float*)d_in[3];
    const float* b0       = (const float*)d_in[4];
    const float* b1       = (const float*)d_in[5];
    const float* att0     = (const float*)d_in[6];
    const float* att1     = (const float*)d_in[7];
    const float* att_b0   = (const float*)d_in[8];
    const float* att_b1   = (const float*)d_in[9];
    const int*   adj_rows = (const int*)d_in[10];
    const int*   adj_cols = (const int*)d_in[11];
    float*       out      = (float*)d_out;

    const int N = in_sizes[0] / DIN;
    const int E = in_sizes[1];

    // workspace layout
    char* ws = (char*)d_ws;
    __bf16* w0pk   = (__bf16*)ws;                                  // 64 KiB
    __bf16* w1pk   = (__bf16*)(ws + WPK_ELEMS * sizeof(__bf16));   // 64 KiB
    float* vw_self  = (float*)(ws + 2 * WPK_ELEMS * sizeof(__bf16));
    float* vw_neigh = vw_self  + (size_t)N * DOUT;
    float* s_self   = vw_neigh + (size_t)N * DOUT;
    float* s_neigh  = s_self   + N;

    // 0) pack weights to bf16 WMMA B fragments
    pack_w_kernel<<<(2 * WPK_ELEMS) / 256, 256, 0, stream>>>(W0, W1, w0pk, w1pk);

    // 1) dual GEMM (WMMA bf16 -> f32)
    gemm_dual_kernel<<<(N + 15) / 16, 256, 0, stream>>>(
        vecs, w0pk, w1pk, vw_self, vw_neigh, N);

    // 2) per-node attention scores
    score_kernel<<<(N + 7) / 8, 256, 0, stream>>>(
        vw_neigh, att0, att1, att_b0, att_b1, s_self, s_neigh, N);

    // 3) per-row softmax aggregation + epilogue
    aggregate_kernel<<<N, 128, 0, stream>>>(
        adj_rows, adj_cols, adj_vals, s_self, s_neigh,
        vw_neigh, vw_self, b0, b1, out, E);
}
